// TorchGemmA8W8_55430847922333
// MI455X (gfx1250) — 8-bit matrix operand order repaired and checked by reading, compiled, not run on hardware
//
#include <hip/hip_runtime.h>
#include <stdint.h>

typedef int          v8i  __attribute__((ext_vector_type(8)));
typedef int          v4i  __attribute__((ext_vector_type(4)));
typedef unsigned int v4u  __attribute__((ext_vector_type(4)));

constexpr int Mdim = 4096;
constexpr int Ndim = 8192;
constexpr int Kdim = 4096;

constexpr int BT   = 128;
constexpr int KT   = 128;
constexpr int LDSS = 144;

union Frag {
    v8i      v;
    uint64_t q[4];
    uint4    x[2];
};

__global__ __launch_bounds__(256) void pack_i32_to_i8(const int* __restrict__ src,
                                                      uint8_t* __restrict__ dst,
                                                      int n4) {
    int i = blockIdx.x * blockDim.x + threadIdx.x;
    if (i < n4) {
        const int4 v = ((const int4*)src)[i];
        uchar4 o;
        o.x = (unsigned char)v.x;
        o.y = (unsigned char)v.y;
        o.z = (unsigned char)v.z;
        o.w = (unsigned char)v.w;
        ((uchar4*)dst)[i] = o;
    }
}

__device__ __forceinline__ void tdm_load_tile(uint32_t lds_off,
                                              const uint8_t* gptr,
                                              uint32_t tensor_rows) {
    const uint64_t ga      = (uint64_t)(uintptr_t)gptr;
    const uint32_t TD0     = Kdim / 4;
    const uint32_t TILE0   = KT / 4;
    const uint32_t TILE1   = BT;
    const uint32_t STRIDE0 = Kdim / 4;

    v4u g0;
    g0[0] = 1u;
    g0[1] = lds_off;
    g0[2] = (uint32_t)ga;
    g0[3] = (uint32_t)((ga >> 32) & 0x01FFFFFFu) | (2u << 30);

    v8i g1;
    g1[0] = (int)((2u << 16)
                | (1u << 20)
                | (4u << 22)
                | (3u << 25));
    g1[1] = (int)((TD0 & 0xFFFFu) << 16);
    g1[2] = (int)((TD0 >> 16) | ((tensor_rows & 0xFFFFu) << 16));
    g1[3] = (int)((tensor_rows >> 16) | (TILE0 << 16));
    g1[4] = (int)(TILE1 & 0xFFFFu);
    g1[5] = (int)STRIDE0;
    g1[6] = 0;
    g1[7] = 0;

    const v4i zz = {0, 0, 0, 0};
#if defined(__clang_major__) && (__clang_major__ >= 23)
    const v8i zz8 = {0, 0, 0, 0, 0, 0, 0, 0};
    __builtin_amdgcn_tensor_load_to_lds(g0, g1, zz, zz, zz8, 0);
#else
    __builtin_amdgcn_tensor_load_to_lds(g0, g1, zz, zz, 0);
#endif
}

__global__ __launch_bounds__(256) void gemm_a8w8_wmma(
        const uint8_t* __restrict__ A,
        const uint8_t* __restrict__ B,
        const _Float16* __restrict__ arow,
        const _Float16* __restrict__ acol,
        _Float16* __restrict__ out) {
    constexpr int TILEB = BT * LDSS;
    __shared__ __align__(16) unsigned char smem[2 * 2 * TILEB];

    const int tid   = threadIdx.x;
    const int lane  = tid & 31;
    const int wave  = tid >> 5;
    const int waveM = wave & 1;
    const int waveN = wave >> 1;
    const int lm    = lane & 15;
    const int half  = lane >> 4;

    const int bm = blockIdx.y * BT;
    const int bn = blockIdx.x * BT;

    v8i acc[4][2];
    const v8i vzero = {0, 0, 0, 0, 0, 0, 0, 0};
#pragma unroll
    for (int mt = 0; mt < 4; ++mt)
#pragma unroll
        for (int nt = 0; nt < 2; ++nt)
            acc[mt][nt] = vzero;

    const uint8_t* aBase = A + (size_t)bm * Kdim;
    const uint8_t* bBase = B + (size_t)bn * Kdim;
    const uint32_t smemOff = (uint32_t)(uintptr_t)&smem[0];

    constexpr int nIter = Kdim / KT;

    if (wave == 0) {
        tdm_load_tile(smemOff + 0 * 2 * TILEB + 0,     aBase + 0, (uint32_t)Mdim);
        tdm_load_tile(smemOff + 0 * 2 * TILEB + TILEB, bBase + 0, (uint32_t)Ndim);
    }

    for (int it = 0; it < nIter; ++it) {
        const int p  = it & 1;
        const int kt = it * KT;

        if (wave == 0) {
            __builtin_amdgcn_s_wait_tensorcnt(0);
        }
        __syncthreads();

        if (it + 1 < nIter && wave == 0) {
            const uint32_t nb = smemOff + (uint32_t)((1 - p) * 2 * TILEB);
            tdm_load_tile(nb,         aBase + kt + KT, (uint32_t)Mdim);
            tdm_load_tile(nb + TILEB, bBase + kt + KT, (uint32_t)Ndim);
        }

        const unsigned char* sA = &smem[p * 2 * TILEB];
        const unsigned char* sB = sA + TILEB;

#pragma unroll
        for (int kk = 0; kk < 2; ++kk) {
            Frag bF[2];
#pragma unroll
            for (int nt = 0; nt < 2; ++nt) {
                const int br = waveN * 32 + nt * 16 + lm;
                bF[nt].x[0] = *(const uint4*)&sB[br * LDSS + kk * 64 + half * 16];
                bF[nt].x[1] = *(const uint4*)&sB[br * LDSS + kk * 64 + half * 16 + 32];
            }
#pragma unroll
            for (int mt = 0; mt < 4; ++mt) {
                Frag aF;
                const int ar = waveM * 64 + mt * 16 + lm;
#pragma unroll
                for (int j = 0; j < 4; ++j)
                    aF.q[j] = *(const uint64_t*)&sA[ar * LDSS + kk * 64 + (j >> 1) * 32 + (j & 1) * 8 + half * 16];
#pragma unroll
                for (int nt = 0; nt < 2; ++nt)
                    acc[mt][nt] = __builtin_amdgcn_wmma_i32_16x16x64_iu8(
                        true, aF.v, true, bF[nt].v, acc[mt][nt],
                         false,  false);
            }
        }
        __syncthreads();
    }

#pragma unroll
    for (int mt = 0; mt < 4; ++mt) {
#pragma unroll
        for (int nt = 0; nt < 2; ++nt) {
            const int n = bn + waveN * 32 + nt * 16 + lm;
            const float ac = (float)acol[n];
#pragma unroll
            for (int g = 0; g < 8; ++g) {
                const int m = bm + waveM * 64 + mt * 16 + half * 8 + g;
                const float ar = (float)arow[m];
                out[(size_t)m * Ndim + n] =
                    (_Float16)((float)acc[mt][nt][g] * ar * ac);
            }
        }
    }
}

extern "C" void kernel_launch(void* const* d_in, const int* in_sizes, int n_in,
                              void* d_out, int out_size, void* d_ws, size_t ws_size,
                              hipStream_t stream) {
    (void)in_sizes; (void)n_in; (void)out_size; (void)ws_size;

    const int*      a32  = (const int*)d_in[0];
    const int*      b32  = (const int*)d_in[1];
    const _Float16* arow = (const _Float16*)d_in[2];
    const _Float16* acol = (const _Float16*)d_in[3];
    _Float16*       out  = (_Float16*)d_out;

    uint8_t* aP = (uint8_t*)d_ws;
    uint8_t* bP = aP + (size_t)Mdim * Kdim;

    {
        const int n4 = (Mdim * Kdim) / 4;
        pack_i32_to_i8<<<(n4 + 255) / 256, 256, 0, stream>>>(a32, aP, n4);
    }
    {
        const int n4 = (Ndim * Kdim) / 4;
        pack_i32_to_i8<<<(n4 + 255) / 256, 256, 0, stream>>>(b32, bP, n4);
    }

    dim3 grid(Ndim / BT, Mdim / BT);
    gemm_a8w8_wmma<<<grid, 256, 0, stream>>>(aP, bP, arow, acol, out);
}
